// GCNClassifier_88648124990825
// MI455X (gfx1250) — matrix operand order repaired and checked by reading, not compiled, not run on hardware
//
#include <hip/hip_runtime.h>
#include <hip/hip_bf16.h>

typedef __attribute__((ext_vector_type(16))) __bf16 v16bf;
typedef __attribute__((ext_vector_type(8)))  float  v8f;

#define EMBED   64
#define HIDDEN  128
#define NCLS    16

// ---------- helpers ----------

// Native f32 -> bf16 conversion (backend emits v_cvt_*_bf16_f32, RNE).
__device__ __forceinline__ __bf16 f2bf(float x) { return (__bf16)x; }

// A fragment (16x32 bf16, MxK) gathered through an index table (embedding lookup).
// Per ISA 7.12.2: lane<16 holds K = {0..7, 16..23}; lane>=16 holds K = {8..15, 24..31}.
__device__ __forceinline__ v16bf a_frag_gather(const float* __restrict__ tab,
                                               const int* __restrict__ ids,
                                               int rowbase, int lane, int kt) {
  int node = ids[rowbase + (lane & 15)];
  const float* rp = tab + (size_t)node * EMBED + kt * 32 + ((lane >> 4) & 1) * 8;
  v16bf a;
#pragma unroll
  for (int j = 0; j < 8; ++j) a[j] = f2bf(rp[j]);
#pragma unroll
  for (int j = 0; j < 8; ++j) a[8 + j] = f2bf(rp[16 + j]);
  return a;
}

// A fragment from a dense row-major f32 activation matrix (leading dim ld).
__device__ __forceinline__ v16bf a_frag_rows(const float* __restrict__ x,
                                             int rowbase, int lane, int kt, int ld) {
  const float* rp = x + (size_t)(rowbase + (lane & 15)) * ld + kt * 32 + ((lane >> 4) & 1) * 8;
  v16bf a;
#pragma unroll
  for (int j = 0; j < 8; ++j) a[j] = f2bf(rp[j]);
#pragma unroll
  for (int j = 0; j < 8; ++j) a[8 + j] = f2bf(rp[16 + j]);
  return a;
}

// B fragment from pre-packed per-lane layout: packed[((nt*KT+kt)*32+lane)*16 + e].
__device__ __forceinline__ v16bf b_frag(const __bf16* __restrict__ packed,
                                        int nt, int kt, int KT, int lane) {
  const __bf16* p = packed + ((size_t)((nt * KT + kt) * 32 + lane)) * 16;
  v16bf b;
#pragma unroll
  for (int j = 0; j < 16; ++j) b[j] = p[j];
  return b;
}

// ---------- kernels ----------

__global__ void k_fill(float* __restrict__ p, long n, float v) {
  long i = (long)blockIdx.x * blockDim.x + threadIdx.x;
  long stride = (long)gridDim.x * blockDim.x;
  for (; i < n; i += stride) p[i] = v;
}

__global__ void k_deg(const int* __restrict__ dst, float* __restrict__ deg, int E) {
  int i = blockIdx.x * blockDim.x + threadIdx.x;
  if (i < E) atomicAdd(deg + dst[i], 1.0f);
}

__global__ void k_dinv(float* __restrict__ deg, int N) {
  int i = blockIdx.x * blockDim.x + threadIdx.x;
  if (i < N) {
    float d = deg[i];
    d = d < 1.0f ? 1.0f : d;
    deg[i] = rsqrtf(d);
  }
}

// Pack f32 weight matrix W[K][Ncols] (row-major) into bf16 B-matrix fragments.
// B layout (32x16 K-major tile): lane<16 -> K=kt*32+{0..7,16..23}; lane>=16 -> K=kt*32+{8..15,24..31}.
__global__ void k_pack_b(const float* __restrict__ W, __bf16* __restrict__ out,
                         int K, int Ncols) {
  int tid = blockIdx.x * blockDim.x + threadIdx.x;
  int KT = K >> 5, NT = Ncols >> 4;
  int total = NT * KT * 32;
  if (tid >= total) return;
  int lane = tid & 31;
  int frag = tid >> 5;
  int kt = frag % KT;
  int nt = frag / KT;
  int n = nt * 16 + (lane & 15);
  int kbase = kt * 32 + ((lane >> 4) & 1) * 8;
  __bf16* o = out + (size_t)tid * 16;
#pragma unroll
  for (int e = 0; e < 16; ++e)
    o[e] = f2bf(W[(size_t)(kbase + (e < 8 ? e : e + 8)) * Ncols + n]);
}

// h = embed[node_ids] @ W1   (N x 64) @ (64 x 128), no bias (GCN adds bias post-agg).
// One block = 16 rows; wave w computes 16-col tile w.  K = 64 -> 2 WMMA steps.
// Output stored as bf16 (halves propagate gather + store traffic).
__global__ void __launch_bounds__(256)
k_gemm_embed(const float* __restrict__ embed, const int* __restrict__ ids,
             const __bf16* __restrict__ Bp, __bf16* __restrict__ out) {
  int rb = blockIdx.x;
  int nt = threadIdx.x >> 5;
  int lane = threadIdx.x & 31;
  int rowbase = rb * 16;

  v16bf a0 = a_frag_gather(embed, ids, rowbase, lane, 0);
  v16bf a1 = a_frag_gather(embed, ids, rowbase, lane, 1);
  v8f acc = {};
  acc = __builtin_amdgcn_wmma_f32_16x16x32_bf16(false, a0, false, b_frag(Bp, nt, 0, 2, lane),
                                                (short)0, acc, false, false);
  acc = __builtin_amdgcn_wmma_f32_16x16x32_bf16(false, a1, false, b_frag(Bp, nt, 1, 2, lane),
                                                (short)0, acc, false, false);
  int n = nt * 16 + (lane & 15);
#pragma unroll
  for (int r = 0; r < 8; ++r) {
    int m = r + ((lane >> 4) & 1) * 8;
    out[(size_t)(rowbase + m) * HIDDEN + n] = f2bf(acc[r]);
  }
}

// h = x @ W2   (N x 128) @ (128 x 128).  K = 128 -> 4 WMMA steps.  bf16 output.
__global__ void __launch_bounds__(256)
k_gemm_act(const float* __restrict__ x, const __bf16* __restrict__ Bp,
           __bf16* __restrict__ out) {
  int rb = blockIdx.x;
  int nt = threadIdx.x >> 5;
  int lane = threadIdx.x & 31;
  int rowbase = rb * 16;

  v8f acc = {};
#pragma unroll
  for (int kt = 0; kt < 4; ++kt) {
    v16bf a = a_frag_rows(x, rowbase, lane, kt, HIDDEN);
    v16bf b = b_frag(Bp, nt, kt, 4, lane);
    acc = __builtin_amdgcn_wmma_f32_16x16x32_bf16(false, a, false, b,
                                                  (short)0, acc, false, false);
  }
  int n = nt * 16 + (lane & 15);
#pragma unroll
  for (int r = 0; r < 8; ++r) {
    int m = r + ((lane >> 4) & 1) * 8;
    out[(size_t)(rowbase + m) * HIDDEN + n] = f2bf(acc[r]);
  }
}

// out = pooled @ Wout + bout   (8192 x 128) @ (128 x 16).  Wave per 16-row block.
__global__ void __launch_bounds__(256)
k_gemm_out(const float* __restrict__ x, const __bf16* __restrict__ Bp,
           const float* __restrict__ bias, float* __restrict__ out) {
  int wave = (int)((blockIdx.x * blockDim.x + threadIdx.x) >> 5);
  int lane = threadIdx.x & 31;
  int rowbase = wave * 16;

  v8f acc = {};
#pragma unroll
  for (int kt = 0; kt < 4; ++kt) {
    v16bf a = a_frag_rows(x, rowbase, lane, kt, HIDDEN);
    v16bf b = b_frag(Bp, 0, kt, 4, lane);
    acc = __builtin_amdgcn_wmma_f32_16x16x32_bf16(false, a, false, b,
                                                  (short)0, acc, false, false);
  }
  int n = lane & 15;
  float bn = bias[n];
#pragma unroll
  for (int r = 0; r < 8; ++r) {
    int m = r + ((lane >> 4) & 1) * 8;
    out[(size_t)(rowbase + m) * NCLS + n] = acc[r] + bn;
  }
}

// Scatter-add propagate: one wave per (edge or self-loop), lane covers 4 columns.
// h rows are bf16 (8 B/lane gather); accumulation stays f32 via global atomics.
__global__ void k_propagate(const __bf16* __restrict__ h, const float* __restrict__ dinv,
                            const int* __restrict__ src, const int* __restrict__ dst,
                            float* __restrict__ agg, int E, int N) {
  int gw = (int)(((long)blockIdx.x * blockDim.x + threadIdx.x) >> 5);
  int lane = threadIdx.x & 31;
  if (gw >= E + N) return;
  int s, d;
  if (gw < E) { s = src[gw]; d = dst[gw]; }
  else        { s = d = gw - E; }
  float nrm = dinv[s] * dinv[d];
  const __bf16* hp = h + (size_t)s * HIDDEN + lane * 4;
  float v0 = (float)hp[0], v1 = (float)hp[1], v2 = (float)hp[2], v3 = (float)hp[3];
  float* ap = agg + (size_t)d * HIDDEN + lane * 4;
  atomicAdd(ap + 0, v0 * nrm);
  atomicAdd(ap + 1, v1 * nrm);
  atomicAdd(ap + 2, v2 * nrm);
  atomicAdd(ap + 3, v3 * nrm);
}

__global__ void k_bias_relu(float* __restrict__ x, const float* __restrict__ b, long n) {
  long i = (long)blockIdx.x * blockDim.x + threadIdx.x;
  if (i < n) {
    float v = x[i] + b[i & (HIDDEN - 1)];
    x[i] = v > 0.0f ? v : 0.0f;
  }
}

__global__ void k_counts(const int* __restrict__ batch, float* __restrict__ counts, int N) {
  int i = blockIdx.x * blockDim.x + threadIdx.x;
  if (i < N) atomicAdd(counts + batch[i], 1.0f);
}

__global__ void k_pool(const float* __restrict__ x, const int* __restrict__ batch,
                       float* __restrict__ pooled, int N) {
  long t = (long)blockIdx.x * blockDim.x + threadIdx.x;
  int node = (int)(t >> 5);
  int lane = (int)(t & 31);
  if (node >= N) return;
  int g = batch[node];
  float4 v = ((const float4*)(x + (size_t)node * HIDDEN))[lane];
  float* pp = pooled + (size_t)g * HIDDEN + lane * 4;
  atomicAdd(pp + 0, v.x);
  atomicAdd(pp + 1, v.y);
  atomicAdd(pp + 2, v.z);
  atomicAdd(pp + 3, v.w);
}

__global__ void k_pool_div(float* __restrict__ pooled, const float* __restrict__ counts,
                           long n) {
  long i = (long)blockIdx.x * blockDim.x + threadIdx.x;
  if (i < n) {
    float c = counts[i >> 7];
    c = c < 1.0f ? 1.0f : c;
    pooled[i] /= c;
  }
}

// ---------- launch ----------

extern "C" void kernel_launch(void* const* d_in, const int* in_sizes, int n_in,
                              void* d_out, int out_size, void* d_ws, size_t ws_size,
                              hipStream_t stream) {
  const int*   node_ids = (const int*)  d_in[0];
  const int*   edges    = (const int*)  d_in[1];
  const int*   batch    = (const int*)  d_in[2];
  const float* embed    = (const float*)d_in[3];
  const float* W1       = (const float*)d_in[4];
  const float* b1       = (const float*)d_in[5];
  const float* W2       = (const float*)d_in[6];
  const float* b2       = (const float*)d_in[7];
  const float* Wout     = (const float*)d_in[8];
  const float* bout     = (const float*)d_in[9];
  float* out = (float*)d_out;

  const int N  = in_sizes[0];
  const int E  = in_sizes[1] / 2;
  const int NG = out_size / NCLS;
  const int* src = edges;
  const int* dst = edges + E;

  // Workspace layout (d_ws is 256B-aligned from hipMalloc).
  float* ws = (float*)d_ws;
  size_t off = (size_t)((N + 31) & ~31);          // floats, 128B-aligned boundary
  float*  dinv   = ws;                             // N f32
  __bf16* hbuf   = (__bf16*)(ws + off);            // N*HIDDEN bf16
  float*  bufB   = ws + off + (size_t)N * HIDDEN / 2;  // N*HIDDEN f32 (agg/activations)
  float*  pooled = bufB + (size_t)N * HIDDEN;      // NG*HIDDEN f32
  float*  counts = pooled + (size_t)NG * HIDDEN;   // NG f32
  __bf16* W1p    = (__bf16*)(counts + NG);         // (128/16)*(64/32)*32*16  = 8192
  __bf16* W2p    = W1p + 8 * 2 * 32 * 16;          // (128/16)*(128/32)*32*16 = 16384
  __bf16* Woutp  = W2p + 8 * 4 * 32 * 16;          // (16/16)*(128/32)*32*16  = 2048

  // Pack weights into WMMA B-fragment layout (bf16).
  k_pack_b<<<2, 256, 0, stream>>>(W1,   W1p,   EMBED,  HIDDEN);
  k_pack_b<<<4, 256, 0, stream>>>(W2,   W2p,   HIDDEN, HIDDEN);
  k_pack_b<<<1, 128, 0, stream>>>(Wout, Woutp, HIDDEN, NCLS);

  // deg (self-loop = 1) -> dinv = rsqrt(max(deg,1))
  k_fill<<<256, 256, 0, stream>>>(dinv, (long)N, 1.0f);
  k_deg<<<(E + 255) / 256, 256, 0, stream>>>(dst, dinv, E);
  k_dinv<<<(N + 255) / 256, 256, 0, stream>>>(dinv, N);

  const long NH = (long)N * HIDDEN;
  const int totalw = E + N;                               // edges + self-loops
  const int prop_blocks = (int)(((long)totalw * 32 + 255) / 256);

  // Layer 1: h1 = embed[ids] @ W1 ; agg1 = norm-scatter(h1) ; relu(agg1 + b1)
  k_gemm_embed<<<N / 16, 256, 0, stream>>>(embed, node_ids, W1p, hbuf);
  k_fill<<<1024, 256, 0, stream>>>(bufB, NH, 0.0f);
  k_propagate<<<prop_blocks, 256, 0, stream>>>(hbuf, dinv, src, dst, bufB, E, N);
  k_bias_relu<<<(int)((NH + 255) / 256), 256, 0, stream>>>(bufB, b1, NH);

  // Layer 2: h2 = x @ W2 ; agg2 = norm-scatter(h2) ; relu(agg2 + b2)
  k_gemm_act<<<N / 16, 256, 0, stream>>>(bufB, W2p, hbuf);
  k_fill<<<1024, 256, 0, stream>>>(bufB, NH, 0.0f);
  k_propagate<<<prop_blocks, 256, 0, stream>>>(hbuf, dinv, src, dst, bufB, E, N);
  k_bias_relu<<<(int)((NH + 255) / 256), 256, 0, stream>>>(bufB, b2, NH);

  // Global mean pool
  k_fill<<<256, 256, 0, stream>>>(pooled, (long)NG * HIDDEN + NG, 0.0f);
  k_counts<<<(N + 255) / 256, 256, 0, stream>>>(batch, counts, N);
  k_pool<<<(int)(((long)N * 32 + 255) / 256), 256, 0, stream>>>(bufB, batch, pooled, N);
  k_pool_div<<<(int)(((long)NG * HIDDEN + 255) / 256), 256, 0, stream>>>(pooled, counts,
                                                                        (long)NG * HIDDEN);

  // Classifier head
  k_gemm_out<<<NG / 16 / 8, 256, 0, stream>>>(pooled, Woutp, bout, out);
}
